// FastMLP_84052509982983
// MI455X (gfx1250) — matrix operand order repaired and checked by reading, not compiled, not run on hardware
//
#include <hip/hip_runtime.h>

// MI455X / gfx1250: wave32. Transposed formulation:
//   h0^T = relu(W0 @ x^T + b0), h1^T = relu(W1 @ h0^T + b1), out^T = W2 @ h1^T + b2
// A operand = weight tiles (register-resident, built once, with the K-permutation
// pi folded in so layer outputs feed the next layer's B operand by pure register
// placement). Biases ride in the WMMA C operand (per-lane v8f, loop-invariant).
// Main loop: no LDS traffic, no bias adds -- just cvt_pk + pk_max + wmma.
typedef __attribute__((ext_vector_type(16))) _Float16 v16h;
typedef __attribute__((ext_vector_type(8)))  _Float16 v8h;
typedef __attribute__((ext_vector_type(2)))  _Float16 h2v;
typedef __attribute__((ext_vector_type(8)))  float    v8f;
typedef __attribute__((ext_vector_type(4)))  float    v4f;

#define TPB   256
#define WAVES 8

static __device__ __forceinline__ v8f wmma_f16(v16h a, v16h b, v8f c) {
    return __builtin_amdgcn_wmma_f32_16x16x32_f16(false, a, false, b, (short)0, c, false, false);
}

__global__ __launch_bounds__(TPB)
void FastMLP_84052509982983_kernel(const float* __restrict__ x,
                                   const float* __restrict__ W0, const float* __restrict__ b0,
                                   const float* __restrict__ W1, const float* __restrict__ b1,
                                   const float* __restrict__ W2, const float* __restrict__ b2,
                                   float* __restrict__ out, int numTiles)
{
    // LDS only for one-time weight/bias fragment construction.
    __shared__ __align__(32) _Float16 lW0[64 * 32];
    __shared__ __align__(32) _Float16 lW1[64 * 64];
    __shared__ __align__(32) _Float16 lW2[16 * 64];   // W2 zero-padded to 16 rows
    __shared__ float lb0[64], lb1[64];

    const int tid = threadIdx.x;
    for (int i = tid; i < 64 * 32; i += TPB) lW0[i] = (_Float16)W0[i];
    for (int i = tid; i < 64 * 64; i += TPB) lW1[i] = (_Float16)W1[i];
    for (int i = tid; i < 16 * 64; i += TPB) lW2[i] = (i < 4 * 64) ? (_Float16)W2[i] : (_Float16)0.f;
    if (tid < 64) { lb0[tid] = b0[tid]; lb1[tid] = b1[tid]; }
    __syncthreads();

    const int lane = tid & 31;
    const int wv   = tid >> 5;
    const int nlo  = lane & 15;   // sample-within-tile (B/C/D column), or weight row (A)
    const int h    = lane >> 4;   // half-wave

    // ---- A fragments: layer 0 uses contiguous K order (k = 16h + s) ----
    // lane (half h): elems 0..15 = K 16*h..16*h+15 (matches x B fragment)
    v16h aw0[4];
#pragma unroll
    for (int t = 0; t < 4; ++t) {
        v8h lo = *(const v8h*)&lW0[(t * 16 + nlo) * 32 + h * 16];
        v8h hi = *(const v8h*)&lW0[(t * 16 + nlo) * 32 + h * 16 + 8];
#pragma unroll
        for (int i = 0; i < 8; ++i) { aw0[t][i] = lo[i]; aw0[t][8 + i] = hi[i]; }
    }
    // ---- A fragments for layers 1/2: pi-permuted layout ----
    // elems 0..7 = W[row][kb*32 + 8*h .. +8], elems 8..15 = W[row][kb*32 + 16 + 8*h .. +8].
    v16h aw1[4][2], aw2[2];
#pragma unroll
    for (int t = 0; t < 4; ++t)
#pragma unroll
        for (int kb = 0; kb < 2; ++kb)
        {
            v8h lo = *(const v8h*)&lW1[(t * 16 + nlo) * 64 + kb * 32 + h * 8];
            v8h hi = *(const v8h*)&lW1[(t * 16 + nlo) * 64 + kb * 32 + 16 + h * 8];
            for (int i = 0; i < 8; ++i) { aw1[t][kb][i] = lo[i]; aw1[t][kb][8 + i] = hi[i]; }
        }
#pragma unroll
    for (int kb = 0; kb < 2; ++kb) {
        v8h lo = *(const v8h*)&lW2[nlo * 64 + kb * 32 + h * 8];
        v8h hi = *(const v8h*)&lW2[nlo * 64 + kb * 32 + 16 + h * 8];
        for (int i = 0; i < 8; ++i) { aw2[kb][i] = lo[i]; aw2[kb][8 + i] = hi[i]; }
    }

    // ---- per-lane bias as WMMA C operands: C elem r -> channel t*16 + 8h + r ----
    v8f cb0[4], cb1[4], cb2;
#pragma unroll
    for (int t = 0; t < 4; ++t)
#pragma unroll
        for (int r = 0; r < 8; ++r) {
            cb0[t][r] = lb0[t * 16 + 8 * h + r];
            cb1[t][r] = lb1[t * 16 + 8 * h + r];
        }
#pragma unroll
    for (int r = 0; r < 8; ++r)
        cb2[r] = (h == 0 && r < 4) ? b2[r] : 0.f;

    const h2v hz = {(_Float16)0.f, (_Float16)0.f};

    // ---- grid-stride loop over 16-sample tiles, x loads software-pipelined ----
    const int tileStride = gridDim.x * WAVES;
    int tile = blockIdx.x * WAVES + wv;
    if (tile >= numTiles) return;

    // lane reads its sample row, features 16*h .. 16*h+15 (b128 x4)
    const float* xr = x + (size_t)(tile * 16 + nlo) * 32 + h * 16;
    v4f xv0 = *(const v4f*)(xr + 0);
    v4f xv1 = *(const v4f*)(xr + 4);
    v4f xv2 = *(const v4f*)(xr + 8);
    v4f xv3 = *(const v4f*)(xr + 12);

    for (; tile < numTiles; tile += tileStride) {
        // B fragment for layer 0 from x (f32 -> packed f16 pairs)
        v16h bx;
#pragma unroll
        for (int i = 0; i < 4; ++i) {
            bx[i]      = (_Float16)xv0[i];
            bx[4 + i]  = (_Float16)xv1[i];
            bx[8 + i]  = (_Float16)xv2[i];
            bx[12 + i] = (_Float16)xv3[i];
        }

        // prefetch next tile's x while the matrix pipe works
        {
            int nt = tile + tileStride;
            int pf = (nt < numTiles) ? nt : tile;
            const float* nxr = x + (size_t)(pf * 16 + nlo) * 32 + h * 16;
            xv0 = *(const v4f*)(nxr + 0);
            xv1 = *(const v4f*)(nxr + 4);
            xv2 = *(const v4f*)(nxr + 8);
            xv3 = *(const v4f*)(nxr + 12);
        }

        // ---- layer 0: bias in C; relu+pack to f16 pairs ----
        h2v pk0[4][4];
#pragma unroll
        for (int t = 0; t < 4; ++t) {
            v8f c = wmma_f16(aw0[t], bx, cb0[t]);
#pragma unroll
            for (int j = 0; j < 4; ++j) {
                h2v p;
                p[0] = (_Float16)c[2 * j];
                p[1] = (_Float16)c[2 * j + 1];
                pk0[t][j] = __builtin_elementwise_max(p, hz);
            }
        }

        // ---- layer 1: B frags by register placement (pi already in aw1) ----
        v16h bh0, bh1;
#pragma unroll
        for (int j = 0; j < 4; ++j) {
            bh0[2 * j]     = pk0[0][j][0];  bh0[2 * j + 1]     = pk0[0][j][1];
            bh0[8 + 2 * j] = pk0[1][j][0];  bh0[8 + 2 * j + 1] = pk0[1][j][1];
            bh1[2 * j]     = pk0[2][j][0];  bh1[2 * j + 1]     = pk0[2][j][1];
            bh1[8 + 2 * j] = pk0[3][j][0];  bh1[8 + 2 * j + 1] = pk0[3][j][1];
        }
        h2v pk1[4][4];
#pragma unroll
        for (int t = 0; t < 4; ++t) {
            v8f c = wmma_f16(aw1[t][0], bh0, cb1[t]);
            c     = wmma_f16(aw1[t][1], bh1, c);
#pragma unroll
            for (int j = 0; j < 4; ++j) {
                h2v p;
                p[0] = (_Float16)c[2 * j];
                p[1] = (_Float16)c[2 * j + 1];
                pk1[t][j] = __builtin_elementwise_max(p, hz);
            }
        }

        // ---- layer 2: bias (incl. b2) in C ----
        v16h bg0, bg1;
#pragma unroll
        for (int j = 0; j < 4; ++j) {
            bg0[2 * j]     = pk1[0][j][0];  bg0[2 * j + 1]     = pk1[0][j][1];
            bg0[8 + 2 * j] = pk1[1][j][0];  bg0[8 + 2 * j + 1] = pk1[1][j][1];
            bg1[2 * j]     = pk1[2][j][0];  bg1[2 * j + 1]     = pk1[2][j][1];
            bg1[8 + 2 * j] = pk1[3][j][0];  bg1[8 + 2 * j + 1] = pk1[3][j][1];
        }
        v8f c2 = wmma_f16(aw2[0], bg0, cb2);
        c2     = wmma_f16(aw2[1], bg1, c2);

        // out^T C tile: lane=sample (lo half), VGPR r = out channel r (r<4 valid)
        if (h == 0) {
            v4f o;
#pragma unroll
            for (int r = 0; r < 4; ++r) o[r] = c2[r];
            *(v4f*)(out + (size_t)(tile * 16 + nlo) * 4) = o;
        }
    }
}

extern "C" void kernel_launch(void* const* d_in, const int* in_sizes, int n_in,
                              void* d_out, int out_size, void* d_ws, size_t ws_size,
                              hipStream_t stream) {
    const float* x  = (const float*)d_in[0];
    const float* W0 = (const float*)d_in[1];
    const float* b0 = (const float*)d_in[2];
    const float* W1 = (const float*)d_in[3];
    const float* b1 = (const float*)d_in[4];
    const float* W2 = (const float*)d_in[5];
    const float* b2 = (const float*)d_in[6];
    float* out = (float*)d_out;

    const int n = in_sizes[0] / 32;      // rows; N is a multiple of 16
    const int numTiles = n / 16;

    int blocks = 2048;                   // 8 waves/block -> ~8 tiles/wave
    if (blocks * WAVES > numTiles) blocks = (numTiles + WAVES - 1) / WAVES;
    if (blocks < 1) blocks = 1;

    FastMLP_84052509982983_kernel<<<blocks, TPB, 0, stream>>>(x, W0, b0, W1, b1, W2, b2,
                                                              out, numTiles);
}
